// BiasedMHA_42958262894960
// MI455X (gfx1250) — compile-verified, matrix operand order checked by reading, not run on hardware
//
#include <hip/hip_runtime.h>
#include <hip/hip_bf16.h>

// B=8, N=1024, F=256, H=8, D=32.  SCALE = 1/sqrt(32).
typedef __attribute__((ext_vector_type(16))) _Float16 v16h;
typedef __attribute__((ext_vector_type(8)))  float    v8f;

#define WMMA_F16(a, b, c) \
  __builtin_amdgcn_wmma_f32_16x16x32_f16(false, (a), false, (b), (short)0, (c), false, false)

// --- async Global->LDS support (gfx1250 ASYNCcnt path), guarded ------------
#if defined(__has_builtin)
#  if __has_builtin(__builtin_amdgcn_global_load_async_to_lds_b128) && \
      __has_builtin(__builtin_amdgcn_s_wait_asynccnt)
#    define USE_ASYNC_LDS 1
#  endif
#endif
#ifndef USE_ASYNC_LDS
#  define USE_ASYNC_LDS 0
#endif

#if USE_ASYNC_LDS
// Builtin signature (from the compiler diagnostic): arg0 is
// 'int __attribute__((vector_size(16))) __device__ *' (global int4*),
// arg1 the LDS-side int4*, then imm offset and imm cpol.
typedef int b128_t __attribute__((vector_size(16)));
typedef __attribute__((address_space(1))) b128_t* as1_b128;
typedef __attribute__((address_space(3))) b128_t* as3_b128;
#endif

__device__ __forceinline__ void async_copy16(const float* g, float* l) {
#if USE_ASYNC_LDS
  // AS1 from generic via integer cast; AS3 offset = low 32 bits of the
  // generic LDS pointer (aperture occupies bits [63:32]).
  __builtin_amdgcn_global_load_async_to_lds_b128(
      (as1_b128)(unsigned long long)g,
      (as3_b128)(unsigned int)(unsigned long long)l, 0, 0);
#else
  *(float4*)l = *(const float4*)g;
#endif
}
__device__ __forceinline__ void wait_async_le4() {
#if USE_ASYNC_LDS
  __builtin_amdgcn_s_wait_asynccnt(4);
#endif
}
__device__ __forceinline__ void wait_async_le0() {
#if USE_ASYNC_LDS
  __builtin_amdgcn_s_wait_asynccnt(0);
#endif
}

// A/B operand fragment for 16x16x32 f16 WMMA, loaded from a row-major row.
// Per ISA layout (16-bit A 16x32): lane<16 -> K chunks [0..7],[16..23];
// lane>=16 -> K chunks [8..15],[24..31].  base = (lane&16)?8:0.
__device__ __forceinline__ v16h load_frag_f32(const float* __restrict__ row, int base) {
  v16h r;
#pragma unroll
  for (int i = 0; i < 8; ++i) r[i] = (_Float16)row[base + i];
#pragma unroll
  for (int i = 0; i < 8; ++i) r[i + 8] = (_Float16)row[base + 16 + i];
  return r;
}
__device__ __forceinline__ v16h load_frag_f16(const _Float16* __restrict__ row, int base) {
  v16h r;
#pragma unroll
  for (int i = 0; i < 8; ++i) r[i] = row[base + i];
#pragma unroll
  for (int i = 0; i < 8; ++i) r[i + 8] = row[base + 16 + i];
  return r;
}

// ---------------------------------------------------------------------------
// Kernel 1: QKV projection.  X(8192x256,f32) @ W^T(256x256,f32) + b.
// 8 waves/block, each wave owns one 16x16 output tile (block tile 16M x 128N).
// Stores f16 in [b][h][n][d] layout; Q is pre-scaled by 1/sqrt(D).
// ---------------------------------------------------------------------------
__global__ __launch_bounds__(256) void qkv_proj_kernel(
    const float* __restrict__ X,
    const float* __restrict__ Wq, const float* __restrict__ bq,
    const float* __restrict__ Wk, const float* __restrict__ bk,
    const float* __restrict__ Wv, const float* __restrict__ bv,
    _Float16* __restrict__ Qh, _Float16* __restrict__ Kh, _Float16* __restrict__ Vh)
{
  const int lane = threadIdx.x & 31;
  const int wave = threadIdx.x >> 5;
  const int m0 = blockIdx.x * 16;                 // 512 blocks over 8192 rows
  const int j0 = blockIdx.y * 128 + wave * 16;    // feature tile
  const int which = blockIdx.z;                   // 0=Q 1=K 2=V

  const float* W    = (which == 0) ? Wq : (which == 1) ? Wk : Wv;
  const float* bias = (which == 0) ? bq : (which == 1) ? bk : bv;
  _Float16*    dst  = (which == 0) ? Qh : (which == 1) ? Kh : Vh;

  const int lh   = lane >> 4;           // lane half-group
  const int base = lh * 8;
  const int arow = m0 + (lane & 15);    // X row for A operand
  const int brow = j0 + (lane & 15);    // W row for B operand (W^T column)

  v8f acc = {};
#pragma unroll
  for (int kk = 0; kk < 256; kk += 32) {
    v16h a  = load_frag_f32(X + (size_t)arow * 256 + kk, base);
    v16h bm = load_frag_f32(W + (size_t)brow * 256 + kk, base);
    acc = WMMA_F16(a, bm, acc);
  }

  const int   col = j0 + (lane & 15);
  const float bj  = bias[col];
  const float scl = (which == 0) ? 0.17677669529663687f : 1.0f;  // 1/sqrt(32)
  const int   h = col >> 5, d = col & 31;
#pragma unroll
  for (int i = 0; i < 8; ++i) {         // C layout: lane<16 rows 0-7, lane>=16 rows 8-15
    int   m  = m0 + i + lh * 8;
    int   bb = m >> 10, nn = m & 1023;
    float v  = (acc[i] + bj) * scl;
    dst[(((size_t)bb * 8 + h) * 1024 + nn) * 32 + d] = (_Float16)v;
  }
}

// ---------------------------------------------------------------------------
// Kernel 2: flash attention.  Block = (b, 16-q tile); wave = head (H==8).
// The 256 MB attn_bias stream is the HBM roofline term.  Each 32-key chunk
// needs a (16q x 32k x 8h) = 16 KB bias tile (16 contiguous 1 KB segments).
// The tile is DOUBLE-BUFFERED in LDS and filled with async global->LDS b128
// copies (ASYNCcnt), so chunk k+1's HBM latency hides behind chunk k's
// WMMA/softmax work; s_wait_asynccnt(4) + barrier publishes the current tile
// (4 ops of the next tile may stay outstanding; async loads retire in order).
// Per chunk: 2 score WMMAs, bias+mask, online softmax (shfl_xor row
// reductions matching the C-layout), P->LDS relayout to A-operand, V->LDS
// transpose to B-operand, 2 PV WMMAs accumulating in C.
// ---------------------------------------------------------------------------
__global__ __launch_bounds__(256) void attn_kernel(
    const _Float16* __restrict__ Qh, const _Float16* __restrict__ Kh,
    const _Float16* __restrict__ Vh,
    const float* __restrict__ bias, const unsigned char* __restrict__ mask,
    _Float16* __restrict__ attn_out)
{
  __shared__ __align__(16) _Float16     pbuf[8][16][32];     // P tile/wave (8 KB)
  __shared__ __align__(16) _Float16     vbuf[8][32][32];     // V tile/wave (16 KB)
  __shared__ __align__(16) float        bbuf[2][16][32][8];  // bias, 2 bufs (32 KB)
  __shared__ __align__(16) unsigned int mbuf[16][8];         // mask tile (512 B)

  const int tid  = threadIdx.x;
  const int lane = tid & 31;
  const int h    = tid >> 5;                 // wave == head
  const int b    = blockIdx.x >> 6;
  const int q0   = (blockIdx.x & 63) * 16;

  const int lh    = lane >> 4;
  const int kbase = lh * 8;
  const int n     = lane & 15;
  const size_t headoff = ((size_t)b * 8 + h) * 1024 * 32;

  // Q fragment (A operand, 16 q-rows x 32 d), held for the whole k loop.
  v16h qa = load_frag_f16(Qh + headoff + (size_t)(q0 + n) * 32, kbase);

  v8f o0 = {}, o1 = {};                      // output accum, d tiles [0,16) [16,32)
  float mrow[8], lrow[8];
#pragma unroll
  for (int i = 0; i < 8; ++i) { mrow[i] = -3.0e38f; lrow[i] = 0.0f; }

  const float* bias_base = bias + ((size_t)b * 1024 + q0) * 1024 * 8;
  const unsigned char* mask_base = mask + ((size_t)b * 1024 + q0) * 1024;

  // This thread's slice of the cooperative 16 KB bias-tile copy: 4 x 16 B.
  // li = p*256+tid -> q row r = li>>6, 16 B chunk cc = li&63.
  // Prologue: issue chunk 0 into buffer 0.
  {
    const float* bsrc = bias_base;
#pragma unroll
    for (int p = 0; p < 4; ++p) {
      int li = p * 256 + tid, r = li >> 6, cc = li & 63;
      async_copy16(bsrc + (size_t)r * 8192 + cc * 4, (float*)bbuf[0][r] + cc * 4);
    }
  }

  for (int kt = 0; kt < 1024; kt += 32) {
    const int  cur      = (kt >> 5) & 1;
    const bool haveNext = (kt + 32 < 1024);

    // --- issue async fill of the NEXT bias tile into the other buffer ---
    if (haveNext) {
      const float* bsrc = bias_base + (size_t)(kt + 32) * 8;
#pragma unroll
      for (int p = 0; p < 4; ++p) {
        int li = p * 256 + tid, r = li >> 6, cc = li & 63;
        async_copy16(bsrc + (size_t)r * 8192 + cc * 4,
                     (float*)bbuf[cur ^ 1][r] + cc * 4);
      }
#if !USE_ASYNC_LDS
      // fallback path: at least prefetch the chunk after next
      if (kt + 64 < 1024)
        __builtin_prefetch(bias_base + (size_t)(kt + 64) * 8 +
                           (size_t)(tid >> 4) * 8192 + (tid & 15) * 16, 0, 0);
#endif
    }

    // --- cooperative mask tile load: 16 rows x 32 bytes (sync, tiny) ---
    if (tid < 128) {
      int r = tid >> 3, cc = tid & 7;
      mbuf[r][cc] = *(const unsigned int*)(mask_base + (size_t)r * 1024 + kt + cc * 4);
    }
    // --- per-wave V tile (32 keys x 32 d) -> LDS, coalesced row loads ---
    {
      const _Float16* vr = Vh + headoff + (size_t)(kt + lane) * 32;
#pragma unroll
      for (int d = 0; d < 32; d += 8)
        *(float4*)&vbuf[h][lane][d] = *(const float4*)(vr + d);
    }

    // --- scores (registers only, overlaps the LDS fills) ---
    v16h kb0 = load_frag_f16(Kh + headoff + (size_t)(kt + n) * 32, kbase);
    v16h kb1 = load_frag_f16(Kh + headoff + (size_t)(kt + 16 + n) * 32, kbase);
    v8f s0 = {}, s1 = {};
    s0 = WMMA_F16(qa, kb0, s0);
    s1 = WMMA_F16(qa, kb1, s1);

    // --- publish current bias tile: own asyncs done, then cross-wave ---
    if (haveNext) wait_async_le4(); else wait_async_le0();
    __syncthreads();   // bias/mask/V tiles visible to all waves

    // --- bias + mask from LDS (mask column 0 force-unmasked) ---
    float p0[8], p1[8];
#pragma unroll
    for (int i = 0; i < 8; ++i) {
      int qr = i + lh * 8;
      const unsigned char* mr = (const unsigned char*)mbuf[qr];
      float x0 = s0[i] + bbuf[cur][qr][n][h];
      float x1 = s1[i] + bbuf[cur][qr][16 + n][h];
      if (mr[n] && (kt + n != 0)) x0 = -1.0e30f;
      if (mr[16 + n])             x1 = -1.0e30f;   // k >= 16, never column 0
      p0[i] = x0; p1[i] = x1;
    }

    // --- online softmax: row reductions within each 16-lane group ---
#pragma unroll
    for (int i = 0; i < 8; ++i) {
      float mx = fmaxf(p0[i], p1[i]);
      for (int off = 1; off < 16; off <<= 1)
        mx = fmaxf(mx, __shfl_xor(mx, off, 32));
      float mnew = fmaxf(mrow[i], mx);
      float corr = __expf(mrow[i] - mnew);
      mrow[i] = mnew;
      float e0 = __expf(p0[i] - mnew);
      float e1 = __expf(p1[i] - mnew);
      p0[i] = e0; p1[i] = e1;
      float rs = e0 + e1;
      for (int off = 1; off < 16; off <<= 1)
        rs += __shfl_xor(rs, off, 32);
      lrow[i] = lrow[i] * corr + rs;
      o0[i] *= corr;
      o1[i] *= corr;
    }

    // --- stage P (C layout -> row-major 16x32 f16) in LDS ---
#pragma unroll
    for (int i = 0; i < 8; ++i) {
      int r = i + lh * 8;
      pbuf[h][r][n]      = (_Float16)p0[i];
      pbuf[h][r][16 + n] = (_Float16)p1[i];
    }
    __syncthreads();   // P visible

    // P as A operand (16 q x 32 k)
    v16h pa;
    {
      const _Float16* pr = &pbuf[h][n][0];
#pragma unroll
      for (int i = 0; i < 8; ++i) pa[i] = pr[kbase + i];
#pragma unroll
      for (int i = 0; i < 8; ++i) pa[8 + i] = pr[kbase + 16 + i];
    }
    // V as B operands (32 k x 16 d per tile): lane = d column, halves = k
    v16h vb0, vb1;
#pragma unroll
    for (int i = 0; i < 8; ++i) {
      vb0[i]     = vbuf[h][kbase + i][n];
      vb0[8 + i] = vbuf[h][kbase + 16 + i][n];
      vb1[i]     = vbuf[h][kbase + i][16 + n];
      vb1[8 + i] = vbuf[h][kbase + 16 + i][16 + n];
    }
    o0 = WMMA_F16(pa, vb0, o0);
    o1 = WMMA_F16(pa, vb1, o1);
    __syncthreads();   // protect tiles before next iteration's fills
  }

  // --- normalize, store f16 in [b][n][h][d] (== (B,N,F) with f = h*32+d) ---
#pragma unroll
  for (int i = 0; i < 8; ++i) {
    int   q   = q0 + i + lh * 8;
    float inv = 1.0f / lrow[i];
    size_t ro = ((size_t)b * 1024 + q) * 256 + h * 32;
    attn_out[ro + n]      = (_Float16)(o0[i] * inv);
    attn_out[ro + 16 + n] = (_Float16)(o1[i] * inv);
  }
}

// ---------------------------------------------------------------------------
// Kernel 3: output projection.  attn(8192x256,f16) @ Wo^T + bo -> f32 out.
// ---------------------------------------------------------------------------
__global__ __launch_bounds__(256) void out_proj_kernel(
    const _Float16* __restrict__ A,
    const float* __restrict__ Wo, const float* __restrict__ bo,
    float* __restrict__ out)
{
  const int lane = threadIdx.x & 31;
  const int wave = threadIdx.x >> 5;
  const int m0 = blockIdx.x * 16;
  const int j0 = blockIdx.y * 128 + wave * 16;
  const int lh   = lane >> 4;
  const int base = lh * 8;
  const int arow = m0 + (lane & 15);
  const int brow = j0 + (lane & 15);

  v8f acc = {};
#pragma unroll
  for (int kk = 0; kk < 256; kk += 32) {
    v16h a  = load_frag_f16(A + (size_t)arow * 256 + kk, base);
    v16h bm = load_frag_f32(Wo + (size_t)brow * 256 + kk, base);
    acc = WMMA_F16(a, bm, acc);
  }
  const int   col = j0 + (lane & 15);
  const float bj  = bo[col];
#pragma unroll
  for (int i = 0; i < 8; ++i) {
    int m = m0 + i + lh * 8;
    out[(size_t)m * 256 + col] = acc[i] + bj;
  }
}

// ---------------------------------------------------------------------------
extern "C" void kernel_launch(void* const* d_in, const int* in_sizes, int n_in,
                              void* d_out, int out_size, void* d_ws, size_t ws_size,
                              hipStream_t stream) {
  const float*         ndata     = (const float*)d_in[0];
  // d_in[1] = N (scalar), compile-time constant 1024 here
  const float*         attn_bias = (const float*)d_in[2];
  const unsigned char* attn_mask = (const unsigned char*)d_in[3];
  const float* Wq = (const float*)d_in[4];
  const float* bq = (const float*)d_in[5];
  const float* Wk = (const float*)d_in[6];
  const float* bk = (const float*)d_in[7];
  const float* Wv = (const float*)d_in[8];
  const float* bv = (const float*)d_in[9];
  const float* Wo = (const float*)d_in[10];
  const float* bo = (const float*)d_in[11];
  float* out = (float*)d_out;

  // Workspace: Q,K,V (f16, [b][h][n][d]) + attn (f16, [b][n][f]) = 16 MB.
  const size_t per = (size_t)8 * 8 * 1024 * 32;  // 2 Mi halves = 4 MB
  _Float16* Qh = (_Float16*)d_ws;
  _Float16* Kh = Qh + per;
  _Float16* Vh = Kh + per;
  _Float16* Ah = Vh + per;

  qkv_proj_kernel<<<dim3(512, 2, 3), 256, 0, stream>>>(
      ndata, Wq, bq, Wk, bk, Wv, bv, Qh, Kh, Vh);
  attn_kernel<<<dim3(512), 256, 0, stream>>>(
      Qh, Kh, Vh, attn_bias, attn_mask, Ah);
  out_proj_kernel<<<dim3(512, 2), 256, 0, stream>>>(Ah, Wo, bo, out);
}
